// MyViT_2121713845033
// MI455X (gfx1250) — matrix operand order repaired and checked by reading, not compiled, not run on hardware
//
#include <hip/hip_runtime.h>
#include <hip/hip_bf16.h>

// ---------------------------------------------------------------------------
// Types for CDNA5 WMMA / TDM (gfx1250, wave32)
// ---------------------------------------------------------------------------
typedef __attribute__((ext_vector_type(16))) __bf16   v16bf;
typedef __attribute__((ext_vector_type(8)))  __bf16   v8bf;
typedef __attribute__((ext_vector_type(8)))  float    v8f;
typedef __attribute__((ext_vector_type(4)))  unsigned uint4v;
typedef __attribute__((ext_vector_type(8)))  int      int8v;
typedef __attribute__((ext_vector_type(4)))  int      int4v;

#define HIDDEN 256
#define NTOK   50
#define BATCH  256
#define ROWS   (BATCH * NTOK)      // 12800
#define MLPD   1024
#define OUTD   1000
#define OUTDP  1024                // padded so N/64 tiles divide evenly
#define INPUTD 3072

// f32 -> bf16 (round to nearest even), no __bf16 arithmetic needed
__device__ __forceinline__ __bf16 f2bf(float f) {
    union { float f; unsigned u; } v; v.f = f;
    unsigned r = v.u + 0x7FFFu + ((v.u >> 16) & 1u);
    unsigned short h = (unsigned short)(r >> 16);
    return __builtin_bit_cast(__bf16, h);
}

__device__ __forceinline__ float pos_emb_val(int t, int j) {
    float jeff = (float)(j & ~1);
    float ang = (float)t * __expf(-jeff * (9.210340371976184f / 256.0f));
    return (j & 1) ? __cosf(ang) : __sinf(ang);
}

// ---------------------------------------------------------------------------
// Tensor Data Mover: async 1-D copy of 4 KB (512 x 8B) global -> LDS.
// D# per CDNA5 ISA: group0 = {count, lds_addr,
// global_addr, type=2}; group1 = {data_size=8B, tensor_dim0=512,
// tensor_dim1=1, tile_dim0=512, tile_dim1=1, dim0_stride=512}.
// Tracked with TENSORcnt.  6-arg builtin form (clang-23 / therock headers).
// ---------------------------------------------------------------------------
__device__ __forceinline__ void tdm_load_4kb(const __bf16* gsrc, __bf16* ldst) {
    unsigned long long ga = (unsigned long long)(uintptr_t)gsrc;
    unsigned lds = (unsigned)(uintptr_t)ldst;       // low 32 bits = LDS offset
    uint4v g0;
    g0[0] = 1u;                                     // count=1 (valid descriptor)
    g0[1] = lds;                                    // lds_addr
    g0[2] = (unsigned)(ga & 0xFFFFFFFFu);           // global_addr[31:0]
    g0[3] = (unsigned)((ga >> 32) & 0x01FFFFFFu)    // global_addr[56:32]
          | (2u << 30);                             // type = 2 ("image")
    int8v g1 = {};
    g1[0] = 3 << 16;                                // data_size = 8 bytes
    g1[1] = (512 & 0xFFFF) << 16;                   // tensor_dim0 = 512 (lo16)
    g1[2] = 1 << 16;                                // dim0 hi16=0 | tensor_dim1 = 1
    g1[3] = 512 << 16;                              // dim1 hi16=0 | tile_dim0 = 512
    g1[4] = 1;                                      // tile_dim1 = 1, tile_dim2 = 0
    g1[5] = 512;                                    // tensor_dim0_stride lo32
    int4v z4 = {};
    int8v z8 = {};
    __builtin_amdgcn_tensor_load_to_lds(g0, g1, z4, z4, z8, 0);
}

// ---------------------------------------------------------------------------
// Positional-embedding table [NTOK, HIDDEN]
// ---------------------------------------------------------------------------
__global__ void posemb_kernel(float* __restrict__ pe) {
    int t = blockIdx.x, j = threadIdx.x;
    pe[t * HIDDEN + j] = pos_emb_val(t, j);
}

// ---------------------------------------------------------------------------
// Swizzle a row-major f32 weight [K, Nsrc] into WMMA B-fragment order:
//   out[t][ntile][lane][i]  (16 contiguous bf16 per lane per k-step)
// ---------------------------------------------------------------------------
__global__ void swizzle_b_kernel(const float* __restrict__ W,
                                 __bf16* __restrict__ out,
                                 int K, int N, int Nsrc) {
    int idx = blockIdx.x * blockDim.x + threadIdx.x;
    if (idx >= K * N) return;
    int i    = idx & 15;
    int lane = (idx >> 4) & 31;
    int rest = idx >> 9;
    int nt   = rest % (N >> 4);
    int t    = rest / (N >> 4);
    int k = t * 32 + (lane >> 4) * 8 + (i & 7) + ((i >> 3) << 4);
    int n = nt * 16 + (lane & 15);
    out[idx] = f2bf(n < Nsrc ? W[(size_t)k * Nsrc + n] : 0.0f);
}

// ---------------------------------------------------------------------------
// CLS token + positional embedding for token 0
// ---------------------------------------------------------------------------
__global__ void cls_init_kernel(const float* __restrict__ cls_tok,
                                const float* __restrict__ pe,
                                float* __restrict__ X) {
    int b = blockIdx.x, n = threadIdx.x;
    X[(size_t)b * NTOK * HIDDEN + n] = cls_tok[n] + pe[n];
}

// ---------------------------------------------------------------------------
// Patch embedding GEMM: X[b, 1+p, :] = patch(b,p) @ Wmap + bmap + posemb
// 8 waves / block; each wave a 16x64 tile (4 WMMAs per K-step).
// ---------------------------------------------------------------------------
__global__ void patch_embed_kernel(const float* __restrict__ images,
                                   const __bf16* __restrict__ Bsw,
                                   const float* __restrict__ bmap,
                                   const float* __restrict__ pe,
                                   float* __restrict__ X) {
    const int lane = threadIdx.x & 31;
    const int gw   = blockIdx.x * (blockDim.x >> 5) + (threadIdx.x >> 5);
    const int ngrp = HIDDEN >> 6;
    const int mt = gw / ngrp, ng = gw % ngrp;
    const int m0 = mt << 4, n0 = ng << 6;

    const int mrow = m0 + (lane & 15);
    const int bimg = mrow / 49, p = mrow % 49;
    const int pi = p / 7, pj = p % 7;
    const float* imgbase = images + (size_t)bimg * 150528 + pi * 32 * 224 + pj * 32;
    const int akb = (lane >> 4) * 8;
    const int ntile = HIDDEN >> 4;

    v8f acc[4] = {};
    for (int t = 0; t < (INPUTD >> 5); ++t) {
        const int k0  = t << 5;
        const int cch = k0 >> 10;
        const int hh  = (k0 >> 5) & 31;
        const float* rowp = imgbase + cch * 50176 + hh * 224;
        v16bf a;
#pragma unroll
        for (int i = 0; i < 8; ++i) {
            a[i]     = f2bf(rowp[akb + i]);
            a[8 + i] = f2bf(rowp[akb + 16 + i]);
        }
        const __bf16* bp = Bsw + (((size_t)t * ntile + (ng << 2)) * 32 + lane) * 16;
#pragma unroll
        for (int j = 0; j < 4; ++j) {
            v16bf bm = *(const v16bf*)(bp + j * 512);
            acc[j] = __builtin_amdgcn_wmma_f32_16x16x32_bf16(
                         false, a, false, bm, (short)0, acc[j], false, false);
        }
    }

#pragma unroll
    for (int j = 0; j < 4; ++j) {
        int cn = n0 + j * 16 + (lane & 15);
        float bias = bmap[cn];
#pragma unroll
        for (int r = 0; r < 8; ++r) {
            int cm  = m0 + r + 8 * (lane >> 4);
            int bb  = cm / 49, tok = cm % 49 + 1;
            X[((size_t)bb * NTOK + tok) * HIDDEN + cn] =
                acc[j][r] + bias + pe[tok * HIDDEN + cn];
        }
    }
}

// ---------------------------------------------------------------------------
// Block-cooperative bf16 WMMA GEMM, B pre-swizzled + staged to LDS by TDM.
// Block = 128 rows x 64 cols; 8 waves each compute a 16x64 tile; wave 0
// double-buffers the shared 4 KB B panel per K-step with tensor_load_to_lds.
// ---------------------------------------------------------------------------
#define GEMM_STORE_F32      0   // Cf = v
#define GEMM_BIAS_GELU_BF16 1   // Cbf = bf16(gelu(v + bias))
#define GEMM_BIAS_ADDTO_F32 2   // Cf += v + bias        (residual)
#define GEMM_BIAS_F32       3   // Cf = v + (n<nreal ? bias : 0)

__global__ void gemm_bf16_kernel(const __bf16* __restrict__ A,
                                 const __bf16* __restrict__ Bsw,
                                 const float* __restrict__ bias,
                                 float* __restrict__ Cf,
                                 __bf16* __restrict__ Cbf,
                                 int M, int N, int K, int mode, int nreal) {
    __shared__ __align__(16) __bf16 lds_b[2][2048];   // 2 x 4 KB double buffer
    const int lane = threadIdx.x & 31;
    const int wid  = threadIdx.x >> 5;
    const int ng   = blockIdx.x;                      // 64-column group
    const int m0   = blockIdx.y * 128 + wid * 16;
    const int ntile = N >> 4;
    const int T = K >> 5;
    const size_t bstep = (size_t)ntile * 512;         // elements per K-step
    const __bf16* bbase = Bsw + (size_t)(ng * 4) * 512;

    if (wid == 0) tdm_load_4kb(bbase, &lds_b[0][0]);

    const __bf16* arow = A + (size_t)(m0 + (lane & 15)) * K + ((lane >> 4) * 8);
    v8f acc[4] = {};
    for (int t = 0; t < T; ++t) {
        if (wid == 0) {
            if (t + 1 < T) {
                tdm_load_4kb(bbase + (size_t)(t + 1) * bstep, &lds_b[(t + 1) & 1][0]);
                __builtin_amdgcn_s_wait_tensorcnt(1);   // buffer t complete
            } else {
                __builtin_amdgcn_s_wait_tensorcnt(0);
            }
        }
        __syncthreads();

        const __bf16* ap = arow + (t << 5);
        __builtin_prefetch(ap + 128, 0, 3);           // -> global_prefetch_b8
        v8bf alo = *(const v8bf*)ap;
        v8bf ahi = *(const v8bf*)(ap + 16);
        v16bf a = __builtin_shufflevector(alo, ahi,
                      0, 1, 2, 3, 4, 5, 6, 7, 8, 9, 10, 11, 12, 13, 14, 15);
        const __bf16* bp = &lds_b[t & 1][lane * 16];
#pragma unroll
        for (int j = 0; j < 4; ++j) {
            v16bf bm = *(const v16bf*)(bp + j * 512); // ds_load_b128 x2
            acc[j] = __builtin_amdgcn_wmma_f32_16x16x32_bf16(
                         false, a, false, bm, (short)0, acc[j], false, false);
        }
        __syncthreads();                              // buffer reuse fence
    }

#pragma unroll
    for (int j = 0; j < 4; ++j) {
        int cn = ng * 64 + j * 16 + (lane & 15);
#pragma unroll
        for (int r = 0; r < 8; ++r) {
            int cm = m0 + r + 8 * (lane >> 4);
            size_t o = (size_t)cm * N + cn;
            float v = acc[j][r];
            if (mode == GEMM_STORE_F32) {
                Cf[o] = v;
            } else if (mode == GEMM_BIAS_GELU_BF16) {
                float tt = v + bias[cn];
                tt = 0.5f * tt * (1.0f + erff(tt * 0.70710678118654752f));
                Cbf[o] = f2bf(tt);
            } else if (mode == GEMM_BIAS_ADDTO_F32) {
                Cf[o] += v + bias[cn];
            } else { // GEMM_BIAS_F32
                Cf[o] = v + (cn < nreal ? bias[cn] : 0.0f);
            }
        }
    }
}

// ---------------------------------------------------------------------------
// LayerNorm over 256 channels -> bf16.  One wave per row, lane owns 8 chans.
// ---------------------------------------------------------------------------
__global__ void ln_bf16_kernel(const float* __restrict__ X,
                               const float* __restrict__ g,
                               const float* __restrict__ b,
                               __bf16* __restrict__ out, int rows) {
    int wid  = (blockIdx.x * blockDim.x + threadIdx.x) >> 5;
    int lane = threadIdx.x & 31;
    if (wid >= rows) return;
    const float* x = X + (size_t)wid * HIDDEN;
    float v[8], s = 0.0f;
#pragma unroll
    for (int i = 0; i < 8; ++i) { v[i] = x[lane + 32 * i]; s += v[i]; }
#pragma unroll
    for (int off = 16; off; off >>= 1) s += __shfl_xor(s, off, 32);
    float mu = s * (1.0f / HIDDEN);
    float q = 0.0f;
#pragma unroll
    for (int i = 0; i < 8; ++i) { float d = v[i] - mu; q += d * d; }
#pragma unroll
    for (int off = 16; off; off >>= 1) q += __shfl_xor(q, off, 32);
    float inv = rsqrtf(q * (1.0f / HIDDEN) + 1e-5f);
#pragma unroll
    for (int i = 0; i < 8; ++i) {
        int c = lane + 32 * i;
        out[(size_t)wid * HIDDEN + c] = f2bf((v[i] - mu) * inv * g[c] + b[c]);
    }
}

// ---------------------------------------------------------------------------
// Per-(row, head) additive attention logits
// ---------------------------------------------------------------------------
__global__ void att_logits_kernel(const float* __restrict__ h,
                                  const float* __restrict__ a_s,
                                  const float* __restrict__ a_d,
                                  float* __restrict__ al_s,
                                  float* __restrict__ al_d, int rows) {
    int idx = blockIdx.x * blockDim.x + threadIdx.x;   // row*8 + head
    if (idx >= rows * 8) return;
    int row = idx >> 3, head = idx & 7;
    const float* hp = h + (size_t)row * HIDDEN + head * 32;
    float ss = 0.0f, sd = 0.0f;
#pragma unroll
    for (int i = 0; i < 32; ++i) {
        float x = hp[i];
        ss += x * a_s[head * 32 + i];
        sd += x * a_d[head * 32 + i];
    }
    al_s[idx] = ss; al_d[idx] = sd;
}

// ---------------------------------------------------------------------------
// GAT aggregation: one workgroup (256 thr) per (batch, target node).
// ---------------------------------------------------------------------------
__global__ void gat_aggregate_kernel(const float* __restrict__ h,
                                     const float* __restrict__ al_s,
                                     const float* __restrict__ al_d,
                                     const int* __restrict__ esrc,
                                     const int* __restrict__ edst, int E,
                                     const float* __restrict__ bias,
                                     float* __restrict__ X) {
    __shared__ int   s_src[64];
    __shared__ int   s_cnt;
    __shared__ float s_alpha[64 * 8];
    const int b = blockIdx.x / NTOK;
    const int n = blockIdx.x % NTOK;

    if (threadIdx.x == 0) {            // deterministic serial edge scan (E=361)
        int c = 0;
        for (int e = 0; e < E; ++e)
            if (edst[e] == n) s_src[c++] = esrc[e];
        s_src[c++] = n;                // GATConv add_self_loops
        s_cnt = c;
    }
    __syncthreads();
    const int cnt = s_cnt;

    if (threadIdx.x < 8) {             // per-head softmax over incoming edges
        int head = threadIdx.x;
        float ald = al_d[((size_t)b * NTOK + n) * 8 + head];
        float m = -1e30f;
        for (int e = 0; e < cnt; ++e) {
            float t = al_s[((size_t)b * NTOK + s_src[e]) * 8 + head] + ald;
            t = (t > 0.0f) ? t : 0.2f * t;             // LeakyReLU(0.2)
            s_alpha[e * 8 + head] = t;
            m = fmaxf(m, t);
        }
        float sum = 0.0f;
        for (int e = 0; e < cnt; ++e) {
            float ex = __expf(s_alpha[e * 8 + head] - m);
            s_alpha[e * 8 + head] = ex;
            sum += ex;
        }
        float inv = 1.0f / (sum + 1e-16f);
        for (int e = 0; e < cnt; ++e) s_alpha[e * 8 + head] *= inv;
    }
    __syncthreads();

    const int c = threadIdx.x, head = c >> 5;
    float acc = 0.0f;
    for (int e = 0; e < cnt; ++e)
        acc += s_alpha[e * 8 + head] * h[((size_t)b * NTOK + s_src[e]) * HIDDEN + c];
    X[((size_t)b * NTOK + n) * HIDDEN + c] += acc + bias[c];
}

// ---------------------------------------------------------------------------
// Extract CLS rows (token 0) -> bf16 [256,256] for the head GEMM
// ---------------------------------------------------------------------------
__global__ void extract_cls_kernel(const float* __restrict__ X,
                                   __bf16* __restrict__ out) {
    int b = blockIdx.x, c = threadIdx.x;
    out[(size_t)b * HIDDEN + c] = f2bf(X[(size_t)b * NTOK * HIDDEN + c]);
}

// ---------------------------------------------------------------------------
// Row softmax over 1000 classes (logits padded to 1024)
// ---------------------------------------------------------------------------
__global__ void softmax_kernel(const float* __restrict__ logits,
                               float* __restrict__ out) {
    __shared__ float red[256];
    const int row = blockIdx.x, t = threadIdx.x;
    const float* lp = logits + (size_t)row * OUTDP;
    float m = -1e30f;
    for (int j = t; j < OUTD; j += 256) m = fmaxf(m, lp[j]);
    red[t] = m; __syncthreads();
    for (int s = 128; s; s >>= 1) { if (t < s) red[t] = fmaxf(red[t], red[t + s]); __syncthreads(); }
    m = red[0]; __syncthreads();
    float sum = 0.0f;
    for (int j = t; j < OUTD; j += 256) sum += __expf(lp[j] - m);
    red[t] = sum; __syncthreads();
    for (int s = 128; s; s >>= 1) { if (t < s) red[t] += red[t + s]; __syncthreads(); }
    float inv = 1.0f / red[0];
    for (int j = t; j < OUTD; j += 256)
        out[(size_t)row * OUTD + j] = __expf(lp[j] - m) * inv;
}

// ---------------------------------------------------------------------------
// Host-side launcher
// ---------------------------------------------------------------------------
extern "C" void kernel_launch(void* const* d_in, const int* in_sizes, int n_in,
                              void* d_out, int out_size, void* d_ws, size_t ws_size,
                              hipStream_t stream) {
    const float* images  = (const float*)d_in[0];
    const int*   ei      = (const int*)  d_in[1];
    const float* Wmap    = (const float*)d_in[2];
    const float* bmap    = (const float*)d_in[3];
    const float* cls_tok = (const float*)d_in[4];
    const float* Wgat    = (const float*)d_in[5];
    const float* att_src = (const float*)d_in[6];
    const float* att_dst = (const float*)d_in[7];
    const float* bgat    = (const float*)d_in[8];
    const float* ln1_g   = (const float*)d_in[9];
    const float* ln1_b   = (const float*)d_in[10];
    const float* ln2_g   = (const float*)d_in[11];
    const float* ln2_b   = (const float*)d_in[12];
    const float* W1      = (const float*)d_in[13];
    const float* b1      = (const float*)d_in[14];
    const float* W2      = (const float*)d_in[15];
    const float* b2      = (const float*)d_in[16];
    const float* Wout    = (const float*)d_in[17];
    const float* bout    = (const float*)d_in[18];
    const int E = in_sizes[1] / 2;
    const int* esrc = ei;
    const int* edst = ei + E;

    // workspace layout
    char* ws = (char*)d_ws;
    size_t off = 0;
    auto alloc = [&](size_t bytes) { size_t o = off; off = (off + bytes + 255) & ~(size_t)255; return o; };
    __bf16* wmap_sw = (__bf16*)(ws + alloc((size_t)INPUTD * HIDDEN * 2));
    __bf16* wgat_sw = (__bf16*)(ws + alloc((size_t)4 * HIDDEN * HIDDEN * 2));
    __bf16* w1_sw   = (__bf16*)(ws + alloc((size_t)4 * HIDDEN * MLPD * 2));
    __bf16* w2_sw   = (__bf16*)(ws + alloc((size_t)4 * MLPD * HIDDEN * 2));
    __bf16* wout_sw = (__bf16*)(ws + alloc((size_t)HIDDEN * OUTDP * 2));
    float*  X       = (float*) (ws + alloc((size_t)ROWS * HIDDEN * 4));
    __bf16* xn_bf   = (__bf16*)(ws + alloc((size_t)ROWS * HIDDEN * 2));
    float*  hbuf    = (float*) (ws + alloc((size_t)ROWS * HIDDEN * 4));
    __bf16* gact_bf = (__bf16*)(ws + alloc((size_t)ROWS * MLPD * 2));
    float*  als     = (float*) (ws + alloc((size_t)ROWS * 8 * 4));
    float*  ald     = (float*) (ws + alloc((size_t)ROWS * 8 * 4));
    __bf16* x0_bf   = (__bf16*)(ws + alloc((size_t)BATCH * HIDDEN * 2));
    float*  logits  = (float*) (ws + alloc((size_t)BATCH * OUTDP * 4));
    float*  pe      = (float*) (ws + alloc((size_t)NTOK * HIDDEN * 4));
    (void)ws_size; (void)n_in; (void)out_size;

    posemb_kernel<<<NTOK, HIDDEN, 0, stream>>>(pe);

    // weight swizzle+convert (B-fragment order)
    {
        int n = INPUTD * HIDDEN;
        swizzle_b_kernel<<<(n + 255) / 256, 256, 0, stream>>>(Wmap, wmap_sw, INPUTD, HIDDEN, HIDDEN);
        n = HIDDEN * HIDDEN;
        for (int l = 0; l < 4; ++l)
            swizzle_b_kernel<<<(n + 255) / 256, 256, 0, stream>>>(
                Wgat + (size_t)l * n, wgat_sw + (size_t)l * n, HIDDEN, HIDDEN, HIDDEN);
        n = HIDDEN * MLPD;
        for (int l = 0; l < 4; ++l) {
            swizzle_b_kernel<<<(n + 255) / 256, 256, 0, stream>>>(
                W1 + (size_t)l * n, w1_sw + (size_t)l * n, HIDDEN, MLPD, MLPD);
            swizzle_b_kernel<<<(n + 255) / 256, 256, 0, stream>>>(
                W2 + (size_t)l * n, w2_sw + (size_t)l * n, MLPD, HIDDEN, HIDDEN);
        }
        n = HIDDEN * OUTDP;
        swizzle_b_kernel<<<(n + 255) / 256, 256, 0, stream>>>(Wout, wout_sw, HIDDEN, OUTDP, OUTD);
    }

    // embedding
    cls_init_kernel<<<BATCH, HIDDEN, 0, stream>>>(cls_tok, pe, X);
    patch_embed_kernel<<<((BATCH * 49 / 16) * (HIDDEN / 64)) / 8, 256, 0, stream>>>(
        images, wmap_sw, bmap, pe, X);

    // transformer-GAT blocks
    for (int l = 0; l < 4; ++l) {
        ln_bf16_kernel<<<ROWS / 8, 256, 0, stream>>>(X, ln1_g + l * HIDDEN,
                                                     ln1_b + l * HIDDEN, xn_bf, ROWS);
        gemm_bf16_kernel<<<dim3(HIDDEN / 64, ROWS / 128), 256, 0, stream>>>(
            xn_bf, wgat_sw + (size_t)l * HIDDEN * HIDDEN, nullptr,
            hbuf, nullptr, ROWS, HIDDEN, HIDDEN, GEMM_STORE_F32, HIDDEN);
        att_logits_kernel<<<(ROWS * 8 + 255) / 256, 256, 0, stream>>>(
            hbuf, att_src + l * HIDDEN, att_dst + l * HIDDEN, als, ald, ROWS);
        gat_aggregate_kernel<<<ROWS, 256, 0, stream>>>(
            hbuf, als, ald, esrc, edst, E, bgat + l * HIDDEN, X);
        ln_bf16_kernel<<<ROWS / 8, 256, 0, stream>>>(X, ln2_g + l * HIDDEN,
                                                     ln2_b + l * HIDDEN, xn_bf, ROWS);
        gemm_bf16_kernel<<<dim3(MLPD / 64, ROWS / 128), 256, 0, stream>>>(
            xn_bf, w1_sw + (size_t)l * HIDDEN * MLPD, b1 + l * MLPD,
            nullptr, gact_bf, ROWS, MLPD, HIDDEN, GEMM_BIAS_GELU_BF16, MLPD);
        gemm_bf16_kernel<<<dim3(HIDDEN / 64, ROWS / 128), 256, 0, stream>>>(
            gact_bf, w2_sw + (size_t)l * MLPD * HIDDEN, b2 + l * HIDDEN,
            X, nullptr, ROWS, HIDDEN, MLPD, GEMM_BIAS_ADDTO_F32, HIDDEN);
    }

    // classifier head + softmax
    extract_cls_kernel<<<BATCH, HIDDEN, 0, stream>>>(X, x0_bf);
    gemm_bf16_kernel<<<dim3(OUTDP / 64, BATCH / 128), 256, 0, stream>>>(
        x0_bf, wout_sw, bout, logits, nullptr, BATCH, OUTDP, HIDDEN,
        GEMM_BIAS_F32, OUTD);
    softmax_kernel<<<BATCH, 256, 0, stream>>>(logits, (float*)d_out);
}
